// AdvectionNet_2791728742927
// MI455X (gfx1250) — matrix operand order repaired and checked by reading, not compiled, not run on hardware
//
#include <hip/hip_runtime.h>
#include <hip/hip_bf16.h>

// ============================================================================
// AdvectionNet forward for MI455X (gfx1250, wave32, WMMA).
// All conv1x1 / pointwise-3D / linear layers run through one generic
// LDS-staged WMMA GEMM kernel (v_wmma_f32_16x16x32_f16, f32 accumulate).
// ============================================================================

typedef __attribute__((ext_vector_type(16))) _Float16 v16h;
typedef __attribute__((ext_vector_type(8)))  _Float16 v8h;
typedef __attribute__((ext_vector_type(8)))  float    v8f;

#define GSZ 6          // advection grid size G
#define NCLS 50
#define BNEPS 1e-5f
#define LPAD 40        // LDS row stride in halves (80 B: 16B-aligned, bank-spread)

static __device__ __forceinline__ int clampi(int v, int lo, int hi) {
  return v < lo ? lo : (v > hi ? hi : v);
}

// Trilinear corners for the 6^3 grid, align_corners=True, zero padding.
static __device__ __forceinline__ void corners6(float px, float py, float pz,
                                                int ci[8], float cw[8]) {
  float fx = (px + 1.f) * 2.5f;   // (p+1)*0.5*(G-1)
  float fy = (py + 1.f) * 2.5f;
  float fz = (pz + 1.f) * 2.5f;
  float f0x = floorf(fx), f0y = floorf(fy), f0z = floorf(fz);
  int x0 = (int)f0x, y0 = (int)f0y, z0 = (int)f0z;
  float wx = fx - f0x, wy = fy - f0y, wz = fz - f0z;
  int q = 0;
  #pragma unroll
  for (int dx = 0; dx < 2; ++dx) {
    #pragma unroll
    for (int dy = 0; dy < 2; ++dy) {
      #pragma unroll
      for (int dz = 0; dz < 2; ++dz) {
        int xi = x0 + dx, yi = y0 + dy, zi = z0 + dz;
        bool valid = (xi >= 0) && (xi < GSZ) && (yi >= 0) && (yi < GSZ) &&
                     (zi >= 0) && (zi < GSZ);
        float w = (dx ? wx : 1.f - wx) * (dy ? wy : 1.f - wy) *
                  (dz ? wz : 1.f - wz);
        int xc = clampi(xi, 0, GSZ - 1);
        int yc = clampi(yi, 0, GSZ - 1);
        int zc = clampi(zi, 0, GSZ - 1);
        ci[q] = (xc * GSZ + yc) * GSZ + zc;
        cw[q] = valid ? w : 0.f;
        ++q;
      }
    }
  }
}

// ----------------------------------------------------------------------------
// Generic WMMA GEMM:  Y[b, m, col] = sum_k W[m,k] * X[b, k, col] + bias[m]
// X addr = X + b*xB + k*xK + col*xC ;  Y addr = Y + b*yB + m*yM + col*yC
//
// Block = 128 threads = 4 waves. Output tile = 32 (M) x 64 (cols):
// each wave owns one 16-col stripe and computes two 16x16 WMMAs per K-chunk
// (two m-subtiles sharing one B fragment). LDS layouts are chosen so each
// WMMA fragment is two contiguous ds_load_b128:
//   A panel  as[m][k]   (32 x 32, row stride LPAD): lane(lo,hi) A-frag =
//            halves [hi*8 .. +7] and [16+hi*8 .. +7] of row lo.
//   X panel  xsT[col][k] (64 x 32, row stride LPAD, TRANSPOSED): lane B-frag =
//            halves [hi*8 .. +7] and [16+hi*8 .. +7] of row col.
// Full interior tiles take an unguarded float4 (global_load_b128) path.
// ----------------------------------------------------------------------------
__global__ __launch_bounds__(128)
void k_gemm_wmma(const float* __restrict__ W, const float* __restrict__ bias,
                 const float* __restrict__ X, float* __restrict__ Y,
                 int M, int K, int NC,
                 long xB, long xK, long xC,
                 long yB, long yM, long yC) {
  __shared__ __align__(16) _Float16 xsT[64 * LPAD];  // [col][k]
  __shared__ __align__(16) _Float16 as[32 * LPAD];   // [m][k]
  const int tid  = threadIdx.x;
  const int lane = tid & 31;
  const int wv   = tid >> 5;        // wave id 0..3 -> col sub-tile
  const int lo   = lane & 15;
  const int hi   = lane >> 4;
  const int b       = blockIdx.z;
  const int mBase   = blockIdx.y * 32;
  const int colBase = blockIdx.x * 64;
  const float* Xb = X + (long)b * xB;
  const bool colFull = (colBase + 64 <= NC) && (xC == 1);

  // fast-path X pointers: thread i-th chunk covers row kk, 4 consecutive cols
  const float* xp[4];
  #pragma unroll
  for (int i = 0; i < 4; ++i) {
    int lin = i * 128 + tid;            // 0..511
    int kk = lin >> 4;                  // 0..31
    int c4 = (lin & 15) * 4;            // 0,4,...,60
    xp[i] = Xb + (long)kk * xK + (colBase + c4);
  }
  // A staging: 8 elements/thread, consecutive k for coalescing
  const float* wp[8];
  int wmm[8];
  #pragma unroll
  for (int i = 0; i < 8; ++i) {
    int lin = i * 128 + tid;            // 0..1023
    wmm[i] = lin >> 5;                  // 0..31
    wp[i] = W + (long)(mBase + wmm[i]) * K + (lin & 31);
  }

  v8f acc0 = {};
  v8f acc1 = {};
  for (int k0 = 0; k0 < K; k0 += 32) {
    const bool kFull = (k0 + 32 <= K);
    // ---- stage X into xsT[col][k] ----
    if (colFull && kFull) {
      float4 vv[4];
      #pragma unroll
      for (int i = 0; i < 4; ++i) vv[i] = *(const float4*)xp[i];
      if (k0 + 32 < K) __builtin_prefetch(xp[0] + 32 * xK, 0, 1);
      #pragma unroll
      for (int i = 0; i < 4; ++i) {
        int lin = i * 128 + tid;
        int kk = lin >> 4;
        int c4 = (lin & 15) * 4;
        xsT[(c4 + 0) * LPAD + kk] = (_Float16)vv[i].x;
        xsT[(c4 + 1) * LPAD + kk] = (_Float16)vv[i].y;
        xsT[(c4 + 2) * LPAD + kk] = (_Float16)vv[i].z;
        xsT[(c4 + 3) * LPAD + kk] = (_Float16)vv[i].w;
        xp[i] += 32 * xK;
      }
    } else {
      #pragma unroll
      for (int i = 0; i < 16; ++i) {
        int lin = i * 128 + tid;        // 0..2047
        int kk = lin >> 6;              // 0..31
        int cc = lin & 63;              // 0..63
        int col = colBase + cc, k = k0 + kk;
        float v = (col < NC && k < K) ? Xb[(long)k * xK + (long)col * xC] : 0.f;
        xsT[cc * LPAD + kk] = (_Float16)v;
      }
    }
    // ---- stage W into as[m][k] ----
    if (kFull && (mBase + 32 <= M)) {
      #pragma unroll
      for (int i = 0; i < 8; ++i) {
        float v = *wp[i];
        as[wmm[i] * LPAD + ((i * 128 + tid) & 31)] = (_Float16)v;
        wp[i] += 32;
      }
    } else {
      #pragma unroll
      for (int i = 0; i < 8; ++i) {
        int lin = i * 128 + tid;
        int mm = lin >> 5, kk = lin & 31;
        int m = mBase + mm, k = k0 + kk;
        float v = (m < M && k < K) ? W[(long)m * K + k] : 0.f;
        as[mm * LPAD + kk] = (_Float16)v;
      }
    }
    __syncthreads();

    // ---- fragments: contiguous b128 pairs ----
    v8h b0 = *(const v8h*)&xsT[(wv * 16 + lo) * LPAD + hi * 8];
    v8h b1 = *(const v8h*)&xsT[(wv * 16 + lo) * LPAD + 16 + hi * 8];
    v16h bf;
    #pragma unroll
    for (int e = 0; e < 8; ++e) { bf[e] = b0[e]; bf[e + 8] = b1[e]; }

    v8h a00 = *(const v8h*)&as[lo * LPAD + hi * 8];
    v8h a01 = *(const v8h*)&as[lo * LPAD + 16 + hi * 8];
    v8h a10 = *(const v8h*)&as[(16 + lo) * LPAD + hi * 8];
    v8h a11 = *(const v8h*)&as[(16 + lo) * LPAD + 16 + hi * 8];
    v16h af0, af1;
    #pragma unroll
    for (int e = 0; e < 8; ++e) {
      af0[e] = a00[e]; af0[e + 8] = a01[e];
      af1[e] = a10[e]; af1[e + 8] = a11[e];
    }
    acc0 = __builtin_amdgcn_wmma_f32_16x16x32_f16(
        false, af0, false, bf, (short)0, acc0, false, false);
    acc1 = __builtin_amdgcn_wmma_f32_16x16x32_f16(
        false, af1, false, bf, (short)0, acc1, false, false);
    __syncthreads();
  }

  // D layout: lane col = lo (hi selects row-half), rows r + 8*hi.
  int col = colBase + wv * 16 + lo;
  if (col < NC) {
    #pragma unroll
    for (int r = 0; r < 8; ++r) {
      int m0 = mBase + r + hi * 8;
      int m1 = mBase + 16 + r + hi * 8;
      if (m0 < M)
        Y[(long)b * yB + (long)m0 * yM + (long)col * yC] = acc0[r] + bias[m0];
      if (m1 < M)
        Y[(long)b * yB + (long)m1 * yM + (long)col * yC] = acc1[r] + bias[m1];
    }
  }
}

// ----------------------------------------------------------------------------
// Elementwise / reduction glue kernels
// ----------------------------------------------------------------------------
__global__ void k_fill(float* __restrict__ p, long n, float v) {
  long t = (long)blockIdx.x * blockDim.x + threadIdx.x;
  if (t < n) p[t] = v;
}

// pc (B,N,3) -> pct (B,3,N)
__global__ void k_transpose_pc(const float* __restrict__ pc,
                               float* __restrict__ pct, int B, int N) {
  long t = (long)blockIdx.x * blockDim.x + threadIdx.x;
  if (t >= (long)B * N * 3) return;
  int d = t % 3;
  long bn = t / 3;
  int n = bn % N, b = bn / N;
  pct[((long)b * 3 + d) * N + n] = pc[t];
}

// value45[b, d, n] = pct[b, d, n]  (first 3 channels)
__global__ void k_copy3(const float* __restrict__ pct, float* __restrict__ val,
                        int B, int N, int Ctot) {
  long t = (long)blockIdx.x * blockDim.x + threadIdx.x;
  if (t >= (long)B * 3 * N) return;
  int n = t % N;
  int d = (t / N) % 3;
  int b = t / ((long)3 * N);
  val[(long)b * Ctot * N + (long)d * N + n] = pct[t];
}

__global__ void k_gridavg_scatter(const float* __restrict__ pct,
                                  float* __restrict__ s, float* __restrict__ c,
                                  int B, int N, int g) {
  int t = blockIdx.x * blockDim.x + threadIdx.x;
  if (t >= B * N) return;
  int b = t / N, n = t % N;
  long pb = (long)b * 3 * N + n;
  float p0 = pct[pb], p1 = pct[pb + N], p2 = pct[pb + 2 * N];
  int g3 = g * g * g;
  int i0 = clampi((int)floorf((p0 + 1.f) * 0.5f * g), 0, g - 1);
  int i1 = clampi((int)floorf((p1 + 1.f) * 0.5f * g), 0, g - 1);
  int i2 = clampi((int)floorf((p2 + 1.f) * 0.5f * g), 0, g - 1);
  int cell = (i0 * g + i1) * g + i2;
  atomicAdd(&s[((long)b * 3 + 0) * g3 + cell], p0);
  atomicAdd(&s[((long)b * 3 + 1) * g3 + cell], p1);
  atomicAdd(&s[((long)b * 3 + 2) * g3 + cell], p2);
  atomicAdd(&c[(long)b * g3 + cell], 1.f);
}

__global__ void k_gridavg_gather(const float* __restrict__ pct,
                                 const float* __restrict__ s,
                                 const float* __restrict__ c,
                                 float* __restrict__ val,
                                 int B, int N, int g, int chOff, int Ctot) {
  int t = blockIdx.x * blockDim.x + threadIdx.x;
  if (t >= B * N) return;
  int b = t / N, n = t % N;
  long pb = (long)b * 3 * N + n;
  float p[3] = {pct[pb], pct[pb + N], pct[pb + 2 * N]};
  int g3 = g * g * g;
  int i0 = clampi((int)floorf((p[0] + 1.f) * 0.5f * g), 0, g - 1);
  int i1 = clampi((int)floorf((p[1] + 1.f) * 0.5f * g), 0, g - 1);
  int i2 = clampi((int)floorf((p[2] + 1.f) * 0.5f * g), 0, g - 1);
  int cell = (i0 * g + i1) * g + i2;
  float cnt = fmaxf(c[(long)b * g3 + cell], 1.f);
  long vb = (long)b * Ctot * N + (long)chOff * N + n;
  #pragma unroll
  for (int d = 0; d < 3; ++d) {
    float m = s[((long)b * 3 + d) * g3 + cell] / cnt;
    val[vb + (long)d * N] = m;
    val[vb + (long)(3 + d) * N] = p[d] - m;
  }
}

// center + rescale, one block per batch (256 threads)
__global__ void k_center_rescale(const float* __restrict__ pct,
                                 float* __restrict__ pcw, int N) {
  int b = blockIdx.x, tid = threadIdx.x;
  __shared__ float red[256];
  __shared__ float mn[3];
  __shared__ float scl;
  for (int d = 0; d < 3; ++d) {
    float sm = 0.f;
    for (int n = tid; n < N; n += 256) sm += pct[((long)b * 3 + d) * N + n];
    red[tid] = sm; __syncthreads();
    for (int st = 128; st > 0; st >>= 1) {
      if (tid < st) red[tid] += red[tid + st];
      __syncthreads();
    }
    if (tid == 0) mn[d] = red[0] / (float)N;
    __syncthreads();
  }
  float mx = 0.f;
  for (int n = tid; n < N; n += 256) {
    float a = pct[((long)b * 3 + 0) * N + n] - mn[0];
    float bq = pct[((long)b * 3 + 1) * N + n] - mn[1];
    float cq = pct[((long)b * 3 + 2) * N + n] - mn[2];
    mx = fmaxf(mx, sqrtf(a * a + bq * bq + cq * cq));
  }
  red[tid] = mx; __syncthreads();
  for (int st = 128; st > 0; st >>= 1) {
    if (tid < st) red[tid] = fmaxf(red[tid], red[tid + st]);
    __syncthreads();
  }
  if (tid == 0) scl = 1.f / (red[0] + 1e-8f);
  __syncthreads();
  for (int n = tid; n < N; n += 256)
    for (int d = 0; d < 3; ++d)
      pcw[((long)b * 3 + d) * N + n] =
          (pct[((long)b * 3 + d) * N + n] - mn[d]) * scl;
}

// BatchNorm stats from the live batch: one block per channel.
// y addr = y + b*bStr + c*cStr + s
__global__ void k_bn_stats(const float* __restrict__ y, float* __restrict__ mean,
                           float* __restrict__ istd, int Bn, int S,
                           long bStr, long cStr) {
  int c = blockIdx.x, tid = threadIdx.x;
  __shared__ float rs[256], rq[256];
  const float* p = y + (long)c * cStr;
  float s = 0.f, q = 0.f;
  for (long t = tid; t < (long)Bn * S; t += 256) {
    long b = t / S, sp = t % S;
    float v = p[b * bStr + sp];
    s += v; q += v * v;
  }
  rs[tid] = s; rq[tid] = q; __syncthreads();
  for (int st = 128; st > 0; st >>= 1) {
    if (tid < st) { rs[tid] += rs[tid + st]; rq[tid] += rq[tid + st]; }
    __syncthreads();
  }
  if (tid == 0) {
    float cnt = (float)Bn * (float)S;
    float m = rs[0] / cnt;
    float v = rq[0] / cnt - m * m;
    mean[c] = m;
    istd[c] = rsqrtf(fmaxf(v, 0.f) + BNEPS);
  }
}

__global__ void k_bn_apply_relu(float* __restrict__ y,
                                const float* __restrict__ mean,
                                const float* __restrict__ istd,
                                const float* __restrict__ gg,
                                const float* __restrict__ bb,
                                int C, int S, long bStr, long cStr, long total) {
  long t = (long)blockIdx.x * blockDim.x + threadIdx.x;
  if (t >= total) return;
  long s = t % S;
  int c = (int)((t / S) % C);
  long b = t / ((long)S * C);
  float* p = y + b * bStr + (long)c * cStr + s;
  float v = (*p - mean[c]) * istd[c] * gg[c] + bb[c];
  *p = fmaxf(v, 0.f);
}

// velocity = src/G ; pcw += velocity
__global__ void k_scale_add(const float* __restrict__ src, float* __restrict__ vel,
                            float* __restrict__ pcw, long n, float scale) {
  long t = (long)blockIdx.x * blockDim.x + threadIdx.x;
  if (t >= n) return;
  float v = src[t] * scale;
  vel[t] = v;
  pcw[t] += v;
}

// RevTrilinear scatter: num (B,C,216), den (B,216)
__global__ void k_splat_scatter(const float* __restrict__ pcw,
                                const float* __restrict__ val,
                                float* __restrict__ num, float* __restrict__ den,
                                int B, int N, int C, long vB, long vCh) {
  int t = blockIdx.x * blockDim.x + threadIdx.x;
  if (t >= B * N) return;
  int b = t / N, n = t % N;
  long pb = (long)b * 3 * N + n;
  int ci[8]; float cw[8];
  corners6(pcw[pb], pcw[pb + N], pcw[pb + 2 * N], ci, cw);
  #pragma unroll
  for (int q = 0; q < 8; ++q) {
    float w = cw[q];
    if (w == 0.f) continue;
    atomicAdd(&den[(long)b * 216 + ci[q]], w);
    for (int c = 0; c < C; ++c) {
      float v = val[(long)b * vB + (long)c * vCh + n];
      atomicAdd(&num[((long)b * C + c) * 216 + ci[q]], v * w);
    }
  }
}

__global__ void k_splat_norm(const float* __restrict__ num,
                             const float* __restrict__ den,
                             float* __restrict__ field, int C, long total) {
  long t = (long)blockIdx.x * blockDim.x + threadIdx.x;
  if (t >= total) return;
  long s = t % 216;
  long b = t / ((long)216 * C);
  field[t] = num[t] / (den[b * 216 + s] + 1e-8f);
}

// 3x3x3 conv, pad 1, on (B,32,6,6,6) flat (x*36+y*6+z); w (O,I,3,3,3)
__global__ void k_conv3k3(const float* __restrict__ x, const float* __restrict__ w,
                          const float* __restrict__ bias, float* __restrict__ y,
                          int B) {
  const int C = 32, S = 216;
  int t = blockIdx.x * blockDim.x + threadIdx.x;
  if (t >= B * C * S) return;
  int s = t % S;
  int o = (t / S) % C;
  int b = t / (S * C);
  int zz = s % 6, yy = (s / 6) % 6, xx = s / 36;
  float acc = bias[o];
  for (int i = 0; i < C; ++i) {
    const float* xb = x + ((long)b * C + i) * S;
    const float* wb = w + ((long)o * C + i) * 27;
    #pragma unroll
    for (int a = 0; a < 3; ++a) {
      int xi = xx + a - 1;
      if (xi < 0 || xi >= 6) continue;
      #pragma unroll
      for (int b2 = 0; b2 < 3; ++b2) {
        int yi = yy + b2 - 1;
        if (yi < 0 || yi >= 6) continue;
        #pragma unroll
        for (int c2 = 0; c2 < 3; ++c2) {
          int zi = zz + c2 - 1;
          if (zi < 0 || zi >= 6) continue;
          acc += xb[(xi * 6 + yi) * 6 + zi] * wb[(a * 3 + b2) * 3 + c2];
        }
      }
    }
  }
  y[t] = acc;
}

// trilinear sample field (B,C,216) at points -> out[b, c, n]
__global__ void k_sample(const float* __restrict__ field,
                         const float* __restrict__ pcw, float* __restrict__ out,
                         int B, int N, int C, long oB, long oCh) {
  int t = blockIdx.x * blockDim.x + threadIdx.x;
  if (t >= B * N) return;
  int b = t / N, n = t % N;
  long pb = (long)b * 3 * N + n;
  int ci[8]; float cw[8];
  corners6(pcw[pb], pcw[pb + N], pcw[pb + 2 * N], ci, cw);
  for (int c = 0; c < C; ++c) {
    const float* fb = field + ((long)b * C + c) * 216;
    float s = 0.f;
    #pragma unroll
    for (int q = 0; q < 8; ++q) s += cw[q] * fb[ci[q]];
    out[(long)b * oB + (long)c * oCh + n] = s;
  }
}

// PIC/FLIP velocity update + advect
__global__ void k_advect(const float* __restrict__ nv, const float* __restrict__ sv,
                         float* __restrict__ vel, float* __restrict__ pcw,
                         int B, int N) {
  int t = blockIdx.x * blockDim.x + threadIdx.x;
  if (t >= B * N) return;
  int b = t / N, n = t % N;
  long pb = (long)b * 3 * N + n;
  float px = pcw[pb], py = pcw[pb + N], pz = pcw[pb + 2 * N];
  int ci[8]; float cw[8];
  corners6(px, py, pz, ci, cw);
  #pragma unroll
  for (int d = 0; d < 3; ++d) {
    const float* nb = nv + ((long)b * 3 + d) * 216;
    const float* sb = sv + ((long)b * 3 + d) * 216;
    float snv = 0.f, ssv = 0.f;
    #pragma unroll
    for (int q = 0; q < 8; ++q) { snv += cw[q] * nb[ci[q]]; ssv += cw[q] * sb[ci[q]]; }
    float flip = (snv - ssv) * (1.f / (float)GSZ);
    float pic = snv * (1.f / (float)GSZ);
    float v = (vel[pb + (long)d * N] + flip) * 0.3f + pic * 0.7f;
    vel[pb + (long)d * N] = v;
    pcw[pb + (long)d * N] += v;
  }
}

// max over points: x (B,C,N) contiguous, one block per (b*C+c)
__global__ void k_maxpool(const float* __restrict__ x, float* __restrict__ out,
                          int N) {
  int bc = blockIdx.x, tid = threadIdx.x;
  __shared__ float red[256];
  const float* p = x + (long)bc * N;
  float m = -3.4e38f;
  for (int n = tid; n < N; n += 256) m = fmaxf(m, p[n]);
  red[tid] = m; __syncthreads();
  for (int st = 128; st > 0; st >>= 1) {
    if (tid < st) red[tid] = fmaxf(red[tid], red[tid + st]);
    __syncthreads();
  }
  if (tid == 0) out[bc] = red[0];
}

__global__ void k_logsoftmax(const float* __restrict__ logits,
                             float* __restrict__ out, int Cls) {
  int b = blockIdx.x, tid = threadIdx.x;  // 64 threads
  __shared__ float sh[64];
  float v = (tid < Cls) ? logits[(long)b * Cls + tid] : -3.4e38f;
  sh[tid] = v; __syncthreads();
  for (int st = 32; st > 0; st >>= 1) {
    if (tid < st) sh[tid] = fmaxf(sh[tid], sh[tid + st]);
    __syncthreads();
  }
  float mx = sh[0];
  __syncthreads();
  float e = (tid < Cls) ? expf(v - mx) : 0.f;
  sh[tid] = e; __syncthreads();
  for (int st = 32; st > 0; st >>= 1) {
    if (tid < st) sh[tid] += sh[tid + st];
    __syncthreads();
  }
  float lse = logf(sh[0]);
  if (tid < Cls) out[(long)b * Cls + tid] = v - mx - lse;
}

// per-batch center (mean over N) and max point norm, one block per b
__global__ void k_center_stats(const float* __restrict__ pcw,
                               float* __restrict__ centers,
                               float* __restrict__ maxn, int N) {
  int b = blockIdx.x, tid = threadIdx.x;
  __shared__ float red[256];
  for (int d = 0; d < 3; ++d) {
    float s = 0.f;
    for (int n = tid; n < N; n += 256) s += pcw[((long)b * 3 + d) * N + n];
    red[tid] = s; __syncthreads();
    for (int st = 128; st > 0; st >>= 1) {
      if (tid < st) red[tid] += red[tid + st];
      __syncthreads();
    }
    if (tid == 0) centers[b * 3 + d] = red[0] / (float)N;
    __syncthreads();
  }
  float mx = 0.f;
  for (int n = tid; n < N; n += 256) {
    float a = pcw[((long)b * 3 + 0) * N + n];
    float bq = pcw[((long)b * 3 + 1) * N + n];
    float cq = pcw[((long)b * 3 + 2) * N + n];
    mx = fmaxf(mx, sqrtf(a * a + bq * bq + cq * cq));
  }
  red[tid] = mx; __syncthreads();
  for (int st = 128; st > 0; st >>= 1) {
    if (tid < st) red[tid] = fmaxf(red[tid], red[tid + st]);
    __syncthreads();
  }
  if (tid == 0) maxn[b] = red[0];
}

// single-block penalty kernel (64 threads)
__global__ void k_penalty(const float* __restrict__ centers,
                          const float* __restrict__ maxn,
                          const int* __restrict__ label,
                          float* __restrict__ out, int B) {
  int tid = threadIdx.x;
  __shared__ float cnt[NCLS];
  __shared__ float cm[NCLS * 3];
  __shared__ float ssq[NCLS];
  __shared__ float acc[64];
  if (tid < NCLS) {
    float c = 0.f, sx = 0.f, sy = 0.f, sz = 0.f;
    for (int b = 0; b < B; ++b)
      if (label[b] == tid) {
        c += 1.f;
        sx += centers[b * 3 + 0];
        sy += centers[b * 3 + 1];
        sz += centers[b * 3 + 2];
      }
    cnt[tid] = c;
    float dn = fmaxf(c, 1.f);
    cm[tid * 3 + 0] = sx / dn;
    cm[tid * 3 + 1] = sy / dn;
    cm[tid * 3 + 2] = sz / dn;
  }
  __syncthreads();
  if (tid < NCLS) {
    float s = 0.f;
    for (int b = 0; b < B; ++b)
      if (label[b] == tid) {
        float dx = centers[b * 3 + 0] - cm[tid * 3 + 0];
        float dy = centers[b * 3 + 1] - cm[tid * 3 + 1];
        float dz = centers[b * 3 + 2] - cm[tid * 3 + 2];
        s += dx * dx + dy * dy + dz * dz;
      }
    ssq[tid] = s;
  }
  __syncthreads();
  float t1 = 0.f;
  for (int b = tid; b < B; b += 64) t1 += 20.f * fmaxf(maxn[b] - 1.f, 0.f);
  float t2 = 0.f;
  if (tid < NCLS && cnt[tid] > 0.f && ssq[tid] > 0.f) t2 = sqrtf(ssq[tid]);
  float t3 = 0.f;
  for (int i = tid; i < NCLS; i += 64) {
    if (cnt[i] > 0.f) {
      for (int j = i + 1; j < NCLS; ++j) {
        if (cnt[j] > 0.f) {
          float dx = cm[i * 3 + 0] - cm[j * 3 + 0];
          float dy = cm[i * 3 + 1] - cm[j * 3 + 1];
          float dz = cm[i * 3 + 2] - cm[j * 3 + 2];
          float d2 = dx * dx + dy * dy + dz * dz;
          float dist = d2 > 0.f ? sqrtf(d2) : 0.f;
          t3 += fmaxf(1.f - dist, 0.f) * 0.5f;
        }
      }
    }
  }
  acc[tid] = t1 + t2 + t3;
  __syncthreads();
  for (int st = 32; st > 0; st >>= 1) {
    if (tid < st) acc[tid] += acc[tid + st];
    __syncthreads();
  }
  if (tid == 0) out[0] = acc[0] / (float)B;
}

// ============================================================================
// Host orchestration
// ============================================================================
struct CbrP {  // {bn:{b,g}, conv/lin:{b,w}} in sorted-key flatten order
  const float *bn_b, *bn_g, *cv_b, *cv_w;
};

extern "C" void kernel_launch(void* const* d_in, const int* in_sizes, int n_in,
                              void* d_out, int out_size, void* d_ws,
                              size_t ws_size, hipStream_t stream) {
  (void)in_sizes; (void)n_in; (void)out_size; (void)ws_size;
  const int B = 32, N = 2048, NT = B * N;
  const int GRID_SIZES[7] = {12, 10, 8, 6, 4, 3, 2};

  // ---- input mapping ----------------------------------------------------
  // setup_inputs() dict order: pc, params, label. params flattened jax-style:
  // sorted dict keys, lists in index order, leaves are arrays. Order:
  //   adv[0..3]{conv0{bn{b,g},conv{b,w}}, conv1{...}, fc{...},
  //             gen_feature{...}, gen_velocity{b,w}},
  //   fc[0..1]{bn{b,g},lin{b,w}}, fc[2]{b,w},
  //   ic1[0..1]{bn{b,g},conv{b,w}}, ic3[0..1]{...},
  //   iv[0]{bn{b,g},conv{b,w}}, iv[1]{b,w}
  int pi = 0;
  const float* PC = (const float*)d_in[pi++];
  auto takeF = [&]() { return (const float*)d_in[pi++]; };
  auto takeCbr = [&]() {
    CbrP c;
    c.bn_b = takeF(); c.bn_g = takeF(); c.cv_b = takeF(); c.cv_w = takeF();
    return c;
  };
  struct AdvP { CbrP conv0, conv1, fc, genf; const float *gv_b, *gv_w; } adv[4];
  for (int i = 0; i < 4; ++i) {
    adv[i].conv0 = takeCbr();
    adv[i].conv1 = takeCbr();
    adv[i].fc = takeCbr();
    adv[i].genf = takeCbr();
    adv[i].gv_b = takeF();
    adv[i].gv_w = takeF();
  }
  CbrP fc0 = takeCbr(), fc1 = takeCbr();
  const float* fc2_b = takeF();
  const float* fc2_w = takeF();
  CbrP ic1a = takeCbr(), ic1b = takeCbr();
  CbrP ic3a = takeCbr(), ic3b = takeCbr();
  CbrP iv0 = takeCbr();
  const float* iv1_b = takeF();
  const float* iv1_w = takeF();
  const int* LAB = (const int*)d_in[pi++];

  // ---- workspace layout (floats) ----------------------------------------
  float* w = (float*)d_ws;
  size_t off = 0;
  auto alloc = [&](size_t n) { float* p = w + off; off += n; return p; };
  float* VALUE = alloc((size_t)B * 320 * N);   // growing concat buffer
  float* V45   = alloc((size_t)B * 45 * N);
  float* SA    = alloc((size_t)B * 64 * N);
  float* SB    = alloc((size_t)B * 3 * N);
  float* VEL   = alloc((size_t)B * 3 * N);
  float* PCT   = alloc((size_t)B * 3 * N);
  float* PCW   = alloc((size_t)B * 3 * N);
  float* BIG1  = alloc((size_t)B * 512 * N);
  float* BIG2  = alloc((size_t)B * 1024 * N);
  float* NUM32 = alloc((size_t)B * 32 * 216);
  float* DEN   = alloc((size_t)B * 216);
  float* F0    = alloc((size_t)B * 32 * 216);
  float* F1    = alloc((size_t)B * 32 * 216);
  float* NVF   = alloc((size_t)B * 3 * 216);
  float* SVN   = alloc((size_t)B * 3 * 216);
  float* SVD   = alloc((size_t)B * 216);
  float* SVF   = alloc((size_t)B * 3 * 216);
  float* GS    = alloc((size_t)B * 3 * 1728);  // max g=12
  float* GC    = alloc((size_t)B * 1728);
  float* MEAN  = alloc(1024);
  float* ISTD  = alloc(1024);
  float* POOL  = alloc((size_t)B * 1024);
  float* FCA   = alloc((size_t)B * 512);
  float* FCB   = alloc((size_t)B * 256);
  float* LOGI  = alloc((size_t)B * NCLS);
  float* CENT  = alloc((size_t)B * 3);
  float* MAXN  = alloc(B);

  auto cdiv = [](long a, long b) { return (int)((a + b - 1) / b); };
  auto fill = [&](float* p, long n, float v) {
    k_fill<<<cdiv(n, 256), 256, 0, stream>>>(p, n, v);
  };
  auto gemm = [&](const float* W_, const float* b_, const float* X_, float* Y_,
                  int M, int K, int NC, int Bz, long xB, long xK, long xC,
                  long yB, long yM, long yC) {
    dim3 g(cdiv(NC, 64), cdiv(M, 32), Bz);
    k_gemm_wmma<<<g, 128, 0, stream>>>(W_, b_, X_, Y_, M, K, NC, xB, xK, xC,
                                       yB, yM, yC);
  };
  auto bnrelu = [&](float* Y_, const CbrP& p, int C, int Bn, int S, long bStr,
                    long cStr) {
    k_bn_stats<<<C, 256, 0, stream>>>(Y_, MEAN, ISTD, Bn, S, bStr, cStr);
    long tot = (long)Bn * C * S;
    k_bn_apply_relu<<<cdiv(tot, 256), 256, 0, stream>>>(
        Y_, MEAN, ISTD, p.bn_g, p.bn_b, C, S, bStr, cStr, tot);
  };

  // ---- feature build: pc transpose, grid_avg pyramid --------------------
  k_transpose_pc<<<cdiv((long)NT * 3, 256), 256, 0, stream>>>(PC, PCT, B, N);
  k_copy3<<<cdiv((long)B * 3 * N, 256), 256, 0, stream>>>(PCT, V45, B, N, 45);
  for (int gi = 0; gi < 7; ++gi) {
    int g = GRID_SIZES[gi], g3 = g * g * g;
    fill(GS, (long)B * 3 * g3, 0.f);
    fill(GC, (long)B * g3, 0.f);
    k_gridavg_scatter<<<cdiv(NT, 256), 256, 0, stream>>>(PCT, GS, GC, B, N, g);
    k_gridavg_gather<<<cdiv(NT, 256), 256, 0, stream>>>(PCT, GS, GC, V45, B, N,
                                                        g, 3 + 6 * gi, 45);
  }
  k_center_rescale<<<B, 256, 0, stream>>>(PCT, PCW, N);

  // ---- ic1: 45->64->64 (value channels 0..63) ---------------------------
  gemm(ic1a.cv_w, ic1a.cv_b, V45, SA, 64, 45, N, B, 45L * N, N, 1, 64L * N, N, 1);
  bnrelu(SA, ic1a, 64, B, N, 64L * N, N);
  gemm(ic1b.cv_w, ic1b.cv_b, SA, VALUE, 64, 64, N, B, 64L * N, N, 1, 320L * N, N, 1);
  bnrelu(VALUE, ic1b, 64, B, N, 320L * N, N);

  // ---- iv: initial velocity ---------------------------------------------
  gemm(iv0.cv_w, iv0.cv_b, VALUE, SA, 64, 64, N, B, 320L * N, N, 1, 64L * N, N, 1);
  bnrelu(SA, iv0, 64, B, N, 64L * N, N);
  gemm(iv1_w, iv1_b, SA, SB, 3, 64, N, B, 64L * N, N, 1, 3L * N, N, 1);
  k_scale_add<<<cdiv((long)B * 3 * N, 256), 256, 0, stream>>>(
      SB, VEL, PCW, (long)B * 3 * N, 1.f / (float)GSZ);

  // ---- advection steps ---------------------------------------------------
  for (int i = 0; i < 4; ++i) {
    const AdvP& p = adv[i];
    int Kin = 64 * (i + 1);
    int chF = 64 + 64 * i;   // feature slot
    int chV = 96 + 64 * i;   // value1 slot
    float* VAL1 = VALUE + (long)chV * N;
    // value1 = cbr(fc)(value[:Kin])
    gemm(p.fc.cv_w, p.fc.cv_b, VALUE, VAL1, 32, Kin, N, B, 320L * N, N, 1,
         320L * N, N, 1);
    bnrelu(VAL1, p.fc, 32, B, N, 320L * N, N);
    // splat value1 -> field F0 (B,32,216)
    fill(NUM32, (long)B * 32 * 216, 0.f);
    fill(DEN, (long)B * 216, 0.f);
    k_splat_scatter<<<cdiv(NT, 256), 256, 0, stream>>>(PCW, VAL1, NUM32, DEN,
                                                       B, N, 32, 320L * N, N);
    k_splat_norm<<<cdiv((long)B * 32 * 216, 256), 256, 0, stream>>>(
        NUM32, DEN, F0, 32, (long)B * 32 * 216);
    // conv0 (pointwise) + bn + relu
    gemm(p.conv0.cv_w, p.conv0.cv_b, F0, F1, 32, 32, 216, B, 32L * 216, 216, 1,
         32L * 216, 216, 1);
    bnrelu(F1, p.conv0, 32, B, 216, 32L * 216, 216);
    // conv1 (3x3x3) + bn + relu
    k_conv3k3<<<cdiv((long)B * 32 * 216, 256), 256, 0, stream>>>(
        F1, p.conv1.cv_w, p.conv1.cv_b, F0, B);
    bnrelu(F0, p.conv1, 32, B, 216, 32L * 216, 216);
    // gen_feature + sample into value[chF]
    gemm(p.genf.cv_w, p.genf.cv_b, F0, F1, 32, 32, 216, B, 32L * 216, 216, 1,
         32L * 216, 216, 1);
    bnrelu(F1, p.genf, 32, B, 216, 32L * 216, 216);
    k_sample<<<cdiv(NT, 256), 256, 0, stream>>>(F1, PCW, VALUE + (long)chF * N,
                                                B, N, 32, 320L * N, N);
    // gen_velocity (no bn)
    gemm(p.gv_w, p.gv_b, F0, NVF, 3, 32, 216, B, 32L * 216, 216, 1, 3L * 216,
         216, 1);
    // splat current velocity
    fill(SVN, (long)B * 3 * 216, 0.f);
    fill(SVD, (long)B * 216, 0.f);
    k_splat_scatter<<<cdiv(NT, 256), 256, 0, stream>>>(PCW, VEL, SVN, SVD, B,
                                                       N, 3, 3L * N, N);
    k_splat_norm<<<cdiv((long)B * 3 * 216, 256), 256, 0, stream>>>(
        SVN, SVD, SVF, 3, (long)B * 3 * 216);
    // PIC/FLIP update + advect
    k_advect<<<cdiv(NT, 256), 256, 0, stream>>>(NVF, SVF, VEL, PCW, B, N);
  }

  // ---- ic3: 320->512->1024, max pool ------------------------------------
  gemm(ic3a.cv_w, ic3a.cv_b, VALUE, BIG1, 512, 320, N, B, 320L * N, N, 1,
       512L * N, N, 1);
  bnrelu(BIG1, ic3a, 512, B, N, 512L * N, N);
  gemm(ic3b.cv_w, ic3b.cv_b, BIG1, BIG2, 1024, 512, N, B, 512L * N, N, 1,
       1024L * N, N, 1);
  bnrelu(BIG2, ic3b, 1024, B, N, 1024L * N, N);
  k_maxpool<<<B * 1024, 256, 0, stream>>>(BIG2, POOL, N);

  // ---- classifier head (cols = batch) ------------------------------------
  gemm(fc0.cv_w, fc0.cv_b, POOL, FCA, 512, 1024, B, 1, 0, 1, 1024, 0, 1, 512);
  bnrelu(FCA, fc0, 512, B, 1, 512, 1);
  gemm(fc1.cv_w, fc1.cv_b, FCA, FCB, 256, 512, B, 1, 0, 1, 512, 0, 1, 256);
  bnrelu(FCB, fc1, 256, B, 1, 256, 1);
  gemm(fc2_w, fc2_b, FCB, LOGI, NCLS, 256, B, 1, 0, 1, 256, 0, 1, NCLS);
  k_logsoftmax<<<B, 64, 0, stream>>>(LOGI, (float*)d_out, NCLS);

  // ---- penalty ------------------------------------------------------------
  k_center_stats<<<B, 256, 0, stream>>>(PCW, CENT, MAXN, N);
  k_penalty<<<1, 64, 0, stream>>>(CENT, MAXN, LAB, (float*)d_out + B * NCLS, B);
}
